// MultiHeadAttention_43559558316843
// MI455X (gfx1250) — matrix operand order repaired and checked by reading, not compiled, not run on hardware
//
#include <hip/hip_runtime.h>
#include <hip/hip_bf16.h>
#include <cstdint>
#include <cstddef>

#define NH    32
#define KVH   8
#define DHEAD 128
#define BATCH 2
#define SEQ   2048
#define EMB   4096
#define NQKV  6144

typedef __attribute__((ext_vector_type(16))) __bf16 v16bf;
typedef __attribute__((ext_vector_type(8)))  float  v8f;

union FragBf {
  uint4 u[2];
  v16bf v;
};

__device__ __forceinline__ v8f zero_v8f() {
  v8f z = {0.f, 0.f, 0.f, 0.f, 0.f, 0.f, 0.f, 0.f};
  return z;
}

__device__ __forceinline__ v8f wmma_bf16(v16bf a, v16bf b, v8f c) {
  // (neg_a, A, neg_b, B, c_mod, C, reuse_a, reuse_b)
  return __builtin_amdgcn_wmma_f32_16x16x32_bf16(false, a, false, b, (short)0, c,
                                                 false, false);
}

// ---------------------------------------------------------------------------
// CDNA5 async global->LDS copies (ASYNCcnt-tracked, no VGPR staging).
// Builtin signature (verified via compiler diagnostic): int4 pointers in the
// global / shared address spaces.
// ---------------------------------------------------------------------------
#if __has_builtin(__builtin_amdgcn_global_load_async_to_lds_b128)
#define HAS_ASYNC 1
#else
#define HAS_ASYNC 0
#endif

typedef int vi4 __attribute__((vector_size(16)));
typedef __attribute__((address_space(1))) vi4* g_vi4p;  // global int4*
typedef __attribute__((address_space(3))) vi4* l_vi4p;  // LDS int4*

__device__ __forceinline__ void cp_async_b128(const void* g, void* l) {
#if HAS_ASYNC
  __builtin_amdgcn_global_load_async_to_lds_b128(
      (g_vi4p)(uintptr_t)g, (l_vi4p)(uint32_t)(uintptr_t)l, 0, 0);
#else
  *(uint4*)l = *(const uint4*)g;
#endif
}

__device__ __forceinline__ void async_wait_all() {
#if HAS_ASYNC
#if __has_builtin(__builtin_amdgcn_s_wait_asynccnt)
  __builtin_amdgcn_s_wait_asynccnt(0);
#else
  asm volatile("s_wait_asynccnt 0x0" ::: "memory");
#endif
#endif
}

// ---------------------------------------------------------------------------
// CDNA5 Tensor Data Mover: one descriptor DMAs a whole 2D tile into LDS
// (TENSORcnt-tracked). D# layout per CDNA5 ISA.
// ROCm 7.2 builtin is 5-arg; the therock/amdgpu-toolchain builtin is 6-arg and
// uniquely ships amd_gfx1250_TDM.h, so __has_include selects the arity.
// ---------------------------------------------------------------------------
#if __has_builtin(__builtin_amdgcn_tensor_load_to_lds)
#define HAS_TDM 1
#if __has_include(<hip/amd_detail/amd_gfx1250_TDM.h>)
#define TDM_SIX_ARGS 1
#else
#define TDM_SIX_ARGS 0
#endif
#else
#define HAS_TDM 0
#endif

typedef unsigned int vu4t __attribute__((vector_size(16)));
typedef int vi8t __attribute__((vector_size(32)));
typedef int vi4t __attribute__((vector_size(16)));

// Load a [tile_h][tile_w] bf16 tile (row stride = stride_elems) into LDS.
// With pad=1, inserts 16B of LDS padding after every 64B row segment, which
// reproduces the __bf16[tile_h][40] layout used by the fragment loaders
// (tile_w must be 32 in that case).
__device__ __forceinline__ void tdm_load_tile_2d(const void* g, void* l,
                                                 int tile_w, int tile_h,
                                                 int stride_elems, int pad) {
#if HAS_TDM
  uint64_t ga = (uint64_t)(uintptr_t)g;
  vu4t g0;
  g0[0] = 1u;                                   // count=1 valid user descriptor
  g0[1] = (uint32_t)(uintptr_t)l;               // lds_addr (bytes)
  g0[2] = (uint32_t)ga;                         // global_addr[31:0]
  g0[3] = (uint32_t)((ga >> 32) & 0x01FFFFFFu)  // global_addr[56:32]
          | (2u << 30);                         // type = 2 ("image")
  uint32_t w0 = (1u << 16);                     // data_size = 1 -> 2 bytes
  if (pad) w0 |= (1u << 20)                     // pad_enable
             |  (3u << 22)                      // pad_interval = 16 DWORDs (64B)
             |  (3u << 25);                     // pad_amount   = 4 DWORDs (16B)
  uint32_t td0 = (uint32_t)stride_elems;        // tensor_dim0 (>= tile_w, in-bounds)
  uint32_t td1 = 0x40000u;                      // tensor_dim1 (large, in-bounds)
  vi8t g1;
  g1[0] = (int)w0;
  g1[1] = (int)((td0 & 0xFFFFu) << 16);                       // dim0[15:0] @63:48
  g1[2] = (int)((td0 >> 16) | ((td1 & 0xFFFFu) << 16));       // dim0[31:16], dim1[15:0]
  g1[3] = (int)((td1 >> 16) | ((uint32_t)tile_w << 16));      // dim1[31:16], tile_dim0
  g1[4] = (int)(uint32_t)tile_h;                              // tile_dim1 (tile_dim2=0)
  g1[5] = (int)(uint32_t)stride_elems;                        // dim0_stride[31:0]
  g1[6] = 0;                                                  // dim0_stride[47:32]
  g1[7] = 0;
  vi4t z4 = {0, 0, 0, 0};
#if TDM_SIX_ARGS
  vi8t z8 = {0, 0, 0, 0, 0, 0, 0, 0};
  __builtin_amdgcn_tensor_load_to_lds(g0, g1, z4, z4, z8, 0);
#else
  __builtin_amdgcn_tensor_load_to_lds(g0, g1, z4, z4, 0);
#endif
#else
  (void)g; (void)l; (void)tile_w; (void)tile_h; (void)stride_elems; (void)pad;
#endif
}

__device__ __forceinline__ void tdm_wait() {
#if HAS_TDM
#if __has_builtin(__builtin_amdgcn_s_wait_tensorcnt)
  __builtin_amdgcn_s_wait_tensorcnt(0);
#else
  asm volatile("s_wait_tensorcnt 0x0" ::: "memory");
#endif
#endif
}

// ---------------------------------------------------------------------------
// Vectorized f32 -> bf16 conversion (n % 1024 == 0 for all our launches).
// ---------------------------------------------------------------------------
__global__ void convert_f32_bf16(const float* __restrict__ src,
                                 __bf16* __restrict__ dst, int n4) {
  int i = blockIdx.x * blockDim.x + threadIdx.x;
  if (i >= n4) return;
  float4 v = *(const float4*)(src + (size_t)i * 4);
  union { __bf16 h[4]; uint2 u; } p;
  p.h[0] = (__bf16)v.x; p.h[1] = (__bf16)v.y;
  p.h[2] = (__bf16)v.z; p.h[3] = (__bf16)v.w;
  *(uint2*)(dst + (size_t)i * 4) = p.u;
}

// ---------------------------------------------------------------------------
// C[M][N] = A[M][K] * Bw[N][K]^T, all-bf16 operands, f32 accumulate.
// Block tile 128x128, K-step 32, 256 threads = 8 waves (4 along M, 2 along N),
// each wave computes 2x4 WMMA 16x16 tiles. Double-buffered LDS, each K-tile
// fetched by a single TDM descriptor per matrix (wave 0 issues, TENSORcnt
// gates), with a per-lane async-to-LDS fallback.
// ---------------------------------------------------------------------------
template <typename OT>
__global__ __launch_bounds__(256)
void gemm_nt_bf16(const __bf16* __restrict__ A, const __bf16* __restrict__ Bw,
                  OT* __restrict__ C, int M, int N, int K) {
  __shared__ __attribute__((aligned(16))) __bf16 As[2][128][40];  // +8 pad
  __shared__ __attribute__((aligned(16))) __bf16 Bs[2][128][40];

  const int tid  = threadIdx.x;
  const int lane = tid & 31;
  const int ln   = lane & 15;
  const int half = lane >> 4;
  const int wid  = tid >> 5;
  const int wm   = wid & 3;   // wave position along M (4 x 32 rows)
  const int wn   = wid >> 2;  // wave position along N (2 x 64 cols)
  const int mb   = blockIdx.y * 128;
  const int nb   = blockIdx.x * 128;

  v8f acc[2][4];
#pragma unroll
  for (int i = 0; i < 2; i++)
#pragma unroll
    for (int j = 0; j < 4; j++) acc[i][j] = zero_v8f();

  auto issue_tile = [&](int buf, int k0) {
#if HAS_TDM
    if (wid == 0) {   // one wave issues; TDM ignores EXEC, issues per wave
      tdm_load_tile_2d(A  + (size_t)mb * K + k0, &As[buf][0][0], 32, 128, K, 1);
      tdm_load_tile_2d(Bw + (size_t)nb * K + k0, &Bs[buf][0][0], 32, 128, K, 1);
    }
#else
    for (int i = tid; i < 512; i += 256) {
      int r = i >> 2, c4 = i & 3;
      cp_async_b128(A  + (size_t)(mb + r) * K + k0 + c4 * 8, &As[buf][r][c4 * 8]);
      cp_async_b128(Bw + (size_t)(nb + r) * K + k0 + c4 * 8, &Bs[buf][r][c4 * 8]);
    }
#endif
  };
  auto wait_tiles = [&]() {
#if HAS_TDM
    if (wid == 0) tdm_wait();
#else
    async_wait_all();
#endif
  };

  issue_tile(0, 0);
  int cur = 0;
  for (int k0 = 0; k0 < K; k0 += 32) {
    wait_tiles();
    __syncthreads();                      // tile[cur] visible to all waves
    if (k0 + 32 < K) issue_tile(cur ^ 1, k0 + 32);  // overlap with compute

    FragBf fa[2], fb[4];
#pragma unroll
    for (int mt = 0; mt < 2; mt++) {
      const __bf16* p = &As[cur][wm * 32 + mt * 16 + ln][half * 8];
      fa[mt].u[0] = *(const uint4*)p;          // K 0-7  / 8-15
      fa[mt].u[1] = *(const uint4*)(p + 16);   // K 16-23 / 24-31
    }
#pragma unroll
    for (int nt = 0; nt < 4; nt++) {
      const __bf16* p = &Bs[cur][wn * 64 + nt * 16 + ln][half * 8];
      fb[nt].u[0] = *(const uint4*)p;          // K 0-7  / 8-15
      fb[nt].u[1] = *(const uint4*)(p + 16);   // K 16-23 / 24-31
    }
#pragma unroll
    for (int mt = 0; mt < 2; mt++)
#pragma unroll
      for (int nt = 0; nt < 4; nt++)
        acc[mt][nt] = wmma_bf16(fa[mt].v, fb[nt].v, acc[mt][nt]);

    __syncthreads();                      // all waves done reading tile[cur]
    cur ^= 1;
  }

  // C layout: VGPR r -> row r (lanes 0-15) or row r+8 (lanes 16-31), col = ln
#pragma unroll
  for (int mt = 0; mt < 2; mt++)
#pragma unroll
    for (int nt = 0; nt < 4; nt++)
#pragma unroll
      for (int r = 0; r < 8; r++) {
        int m = mb + wm * 32 + mt * 16 + r + half * 8;
        int n = nb + wn * 64 + nt * 16 + ln;
        C[(size_t)m * N + n] = (OT)acc[mt][nt][r];
      }
}

// ---------------------------------------------------------------------------
// RoPE on Q and K heads of the fused projection (interleaved-pair convention,
// matching the reference's even/odd split). Writes Q:[B,NH,S,D], K:[B,KVH,S,D].
// ---------------------------------------------------------------------------
__global__ void rope_qk(const __bf16* __restrict__ proj,
                        __bf16* __restrict__ Q, __bf16* __restrict__ Kc) {
  const int PAIRS = DHEAD / 2;
  int i = blockIdx.x * blockDim.x + threadIdx.x;
  const int total = BATCH * SEQ * (NH + KVH) * PAIRS;
  if (i >= total) return;
  int pr = i % PAIRS;
  int h  = (i / PAIRS) % (NH + KVH);
  int s  = (i / (PAIRS * (NH + KVH))) % SEQ;
  int b  = i / (PAIRS * (NH + KVH) * SEQ);

  float inv_freq = __powf(10000.f, -(float)(2 * pr) / (float)DHEAD);
  float ang = (float)s * inv_freq;
  float cs = __cosf(ang), sn = __sinf(ang);

  size_t head_off = (size_t)h * DHEAD;  // q heads then k heads are contiguous
  const __bf16* src = proj + ((size_t)(b * SEQ + s)) * NQKV + head_off;
  float x1 = (float)src[2 * pr];
  float x2 = (float)src[2 * pr + 1];
  float o1 = x1 * cs - x2 * sn;
  float o2 = x1 * sn + x2 * cs;

  __bf16* dst;
  if (h < NH) dst = Q + (((size_t)(b * NH + h)) * SEQ + s) * DHEAD;
  else        dst = Kc + (((size_t)(b * KVH + (h - NH))) * SEQ + s) * DHEAD;
  dst[2 * pr]     = (__bf16)o1;
  dst[2 * pr + 1] = (__bf16)o2;
}

// V from proj -> transposed bf16 layout Vt:[B,KVH,D,S] so the PV B-fragment
// ([n=d][k=kpos], contiguous along kpos) is a straight LDS copy.
__global__ void v_transpose(const __bf16* __restrict__ proj,
                            __bf16* __restrict__ Vt) {
  int i = blockIdx.x * blockDim.x + threadIdx.x;
  const int total = BATCH * KVH * DHEAD * SEQ;
  if (i >= total) return;
  int s = i % SEQ;
  int d = (i / SEQ) % DHEAD;
  int h = (i / (SEQ * DHEAD)) % KVH;
  int b = i / (SEQ * DHEAD * KVH);
  Vt[i] = proj[((size_t)(b * SEQ + s)) * NQKV + (NH + KVH) * DHEAD +
               (size_t)h * DHEAD + d];
}

// ---------------------------------------------------------------------------
// Causal flash attention with GQA (32 q-heads over 8 kv-heads).
// Block = 128 threads = 4 waves; each wave owns 16 q-rows; block covers 64.
// K/V tiles double-buffered via async global->LDS copies. Per 32-key block:
// 8 WMMA QK^T + 8 WMMA PV. Online softmax in f32 with xor-shuffle row
// reductions matching the C-tile VGPR layout.
// ---------------------------------------------------------------------------
__global__ __launch_bounds__(128)
void flash_attn(const __bf16* __restrict__ Q, const __bf16* __restrict__ Kc,
                const __bf16* __restrict__ Vt, __bf16* __restrict__ O) {
  __shared__ __attribute__((aligned(16))) __bf16 Qs[64][136];
  __shared__ __attribute__((aligned(16))) __bf16 Ks[2][32][136];
  __shared__ __attribute__((aligned(16))) __bf16 Vs[2][128][40];
  __shared__ __attribute__((aligned(16))) __bf16 Ps[4][16][40];

  const int tid  = threadIdx.x;
  const int lane = tid & 31;
  const int ln   = lane & 15;
  const int half = lane >> 4;
  const int w    = tid >> 5;
  const int qb   = blockIdx.x * 64;
  const int h    = blockIdx.y;
  const int b    = blockIdx.z;
  const int kvh  = h >> 2;  // NH / KVH == 4

  const __bf16* Qg = Q  + (((size_t)(b * NH + h)) * SEQ + qb) * DHEAD;
  const __bf16* Kg = Kc + ((size_t)(b * KVH + kvh)) * SEQ * DHEAD;
  const __bf16* Vg = Vt + ((size_t)(b * KVH + kvh)) * DHEAD * SEQ;

  auto issue_kv = [&](int buf, int kb) {
    for (int i = tid; i < 512; i += 128) {  // K tile 32x128
      int r = i >> 4, c4 = i & 15;
      cp_async_b128(Kg + (size_t)(kb + r) * DHEAD + c4 * 8, &Ks[buf][r][c4 * 8]);
    }
    for (int i = tid; i < 512; i += 128) {  // Vt tile 128x32
      int r = i >> 2, c4 = i & 3;
      cp_async_b128(Vg + (size_t)r * SEQ + kb + c4 * 8, &Vs[buf][r][c4 * 8]);
    }
  };

  // Q tile (64x128 = 1024 b128) + first K/V tile, one async batch
  for (int i = tid; i < 1024; i += 128) {
    int r = i >> 4, c4 = i & 15;
    cp_async_b128(Qg + (size_t)r * DHEAD + c4 * 8, &Qs[r][c4 * 8]);
  }
  issue_kv(0, 0);
  async_wait_all();
  __syncthreads();

  FragBf fq[4];
#pragma unroll
  for (int c = 0; c < 4; c++) {
    const __bf16* p = &Qs[w * 16 + ln][c * 32 + half * 8];
    fq[c].u[0] = *(const uint4*)p;
    fq[c].u[1] = *(const uint4*)(p + 16);
  }

  v8f o[8];
#pragma unroll
  for (int g = 0; g < 8; g++) o[g] = zero_v8f();
  float run_m[8], run_l[8];
#pragma unroll
  for (int r = 0; r < 8; r++) { run_m[r] = -3.0e38f; run_l[r] = 0.f; }

  const float scale = 0.088388347648318447f;  // 1/sqrt(128)
  const int kend = qb + 64;
  int cur = 0;

  for (int kb = 0; kb < kend; kb += 32) {
    if (kb + 32 < kend) issue_kv(cur ^ 1, kb + 32);  // overlap with compute

    // S = Q * K^T : two 16x16 n-tiles covering 32 key positions
    v8f s0 = zero_v8f(), s1 = zero_v8f();
#pragma unroll
    for (int c = 0; c < 4; c++) {
      FragBf fk0, fk1;
      const __bf16* p0 = &Ks[cur][ln][c * 32 + half * 8];
      fk0.u[0] = *(const uint4*)p0;
      fk0.u[1] = *(const uint4*)(p0 + 16);
      const __bf16* p1 = &Ks[cur][16 + ln][c * 32 + half * 8];
      fk1.u[0] = *(const uint4*)p1;
      fk1.u[1] = *(const uint4*)(p1 + 16);
      s0 = wmma_bf16(fq[c].v, fk0.v, s0);
      s1 = wmma_bf16(fq[c].v, fk1.v, s1);
    }

    // Online softmax. Row m = r + 8*half lives in one VGPR across a 16-lane
    // half, so xor-shuffles 1/2/4/8 reduce exactly one row.
#pragma unroll
    for (int r = 0; r < 8; r++) {
      int m  = qb + w * 16 + r + half * 8;
      int n0 = kb + ln;
      int n1 = kb + 16 + ln;
      bool vv0 = (n0 <= m), vv1 = (n1 <= m);
      float a0 = vv0 ? s0[r] * scale : -3.0e38f;
      float a1 = vv1 ? s1[r] * scale : -3.0e38f;
      float mx = fmaxf(a0, a1);
      mx = fmaxf(mx, __shfl_xor(mx, 1, 32));
      mx = fmaxf(mx, __shfl_xor(mx, 2, 32));
      mx = fmaxf(mx, __shfl_xor(mx, 4, 32));
      mx = fmaxf(mx, __shfl_xor(mx, 8, 32));
      float nm  = fmaxf(run_m[r], mx);
      float p0v = vv0 ? __expf(a0 - nm) : 0.f;
      float p1v = vv1 ? __expf(a1 - nm) : 0.f;
      float alpha = __expf(run_m[r] - nm);
      float rs = p0v + p1v;
      rs += __shfl_xor(rs, 1, 32);
      rs += __shfl_xor(rs, 2, 32);
      rs += __shfl_xor(rs, 4, 32);
      rs += __shfl_xor(rs, 8, 32);
      run_l[r] = run_l[r] * alpha + rs;
      run_m[r] = nm;
#pragma unroll
      for (int g = 0; g < 8; g++) o[g][r] *= alpha;
      // stage P (C layout) into LDS to reload in A-fragment layout
      Ps[w][r + half * 8][ln]      = (__bf16)p0v;
      Ps[w][r + half * 8][16 + ln] = (__bf16)p1v;
    }

    // O += P * V
    FragBf fp;
    {
      const __bf16* p = &Ps[w][ln][half * 8];
      fp.u[0] = *(const uint4*)p;
      fp.u[1] = *(const uint4*)(p + 16);
    }
#pragma unroll
    for (int g = 0; g < 8; g++) {
      FragBf fv;
      const __bf16* p = &Vs[cur][g * 16 + ln][half * 8];
      fv.u[0] = *(const uint4*)p;
      fv.u[1] = *(const uint4*)(p + 16);
      o[g] = wmma_bf16(fp.v, fv.v, o[g]);
    }

    async_wait_all();   // next tile landed
    __syncthreads();    // everyone done reading tile[cur]
    cur ^= 1;
  }

  // normalize and store to [b][s][h][d] (row-major M x 4096 for dense GEMM)
#pragma unroll
  for (int r = 0; r < 8; r++) {
    float inv = 1.f / run_l[r];
    int m = qb + w * 16 + r + half * 8;
    __bf16* dst = O + (((size_t)(b * SEQ + m)) * NH + h) * DHEAD + ln;
#pragma unroll
    for (int g = 0; g < 8; g++) dst[g * 16] = (__bf16)(o[g][r] * inv);
  }
}

// ---------------------------------------------------------------------------
extern "C" void kernel_launch(void* const* d_in, const int* in_sizes, int n_in,
                              void* d_out, int out_size, void* d_ws,
                              size_t ws_size, hipStream_t stream) {
  (void)in_sizes; (void)n_in; (void)out_size; (void)ws_size;
  const float* qkv     = (const float*)d_in[0];
  // d_in[1] = position_ids (arange; positions derived from s directly)
  const float* w_qkv   = (const float*)d_in[2];
  const float* w_dense = (const float*)d_in[3];
  float* out = (float*)d_out;

  char* ws = (char*)d_ws;
  size_t off = 0;
  __bf16* Abf   = (__bf16*)(ws + off); off += (size_t)BATCH * SEQ * EMB * 2;
  __bf16* Wqkvb = (__bf16*)(ws + off); off += (size_t)NQKV * EMB * 2;
  __bf16* Wdb   = (__bf16*)(ws + off); off += (size_t)EMB * EMB * 2;
  __bf16* proj  = (__bf16*)(ws + off); off += (size_t)BATCH * SEQ * NQKV * 2;
  __bf16* Qb    = (__bf16*)(ws + off); off += (size_t)BATCH * NH  * SEQ * DHEAD * 2;
  __bf16* Kb    = (__bf16*)(ws + off); off += (size_t)BATCH * KVH * SEQ * DHEAD * 2;
  __bf16* Vtb   = (__bf16*)(ws + off); off += (size_t)BATCH * KVH * DHEAD * SEQ * 2;
  __bf16* attn  = (__bf16*)(ws + off); off += (size_t)BATCH * SEQ * NH * DHEAD * 2;
  // total workspace use: 240 MiB

  {  // 0) one-time f32 -> bf16 conversions (activations + weights)
    int n4;
    n4 = BATCH * SEQ * EMB / 4;
    convert_f32_bf16<<<(n4 + 255) / 256, 256, 0, stream>>>(qkv, Abf, n4);
    n4 = NQKV * EMB / 4;
    convert_f32_bf16<<<(n4 + 255) / 256, 256, 0, stream>>>(w_qkv, Wqkvb, n4);
    n4 = EMB * EMB / 4;
    convert_f32_bf16<<<(n4 + 255) / 256, 256, 0, stream>>>(w_dense, Wdb, n4);
  }
  {  // 1) fused QKV projection: [4096,4096] x [6144,4096]^T -> bf16
    dim3 grid(NQKV / 128, (BATCH * SEQ) / 128);
    gemm_nt_bf16<__bf16><<<grid, 256, 0, stream>>>(Abf, Wqkvb, proj,
                                                   BATCH * SEQ, NQKV, EMB);
  }
  {  // 2) RoPE on q/k heads
    int total = BATCH * SEQ * (NH + KVH) * (DHEAD / 2);
    rope_qk<<<(total + 255) / 256, 256, 0, stream>>>(proj, Qb, Kb);
  }
  {  // 3) V -> [B,KVH,D,S]
    int total = BATCH * KVH * DHEAD * SEQ;
    v_transpose<<<(total + 255) / 256, 256, 0, stream>>>(proj, Vtb);
  }
  {  // 4) causal GQA flash attention
    dim3 grid(SEQ / 64, NH, BATCH);
    flash_attn<<<grid, 128, 0, stream>>>(Qb, Kb, Vtb, attn);
  }
  {  // 5) output projection: [4096,4096] x [4096,4096]^T -> f32 out
    dim3 grid(EMB / 128, (BATCH * SEQ) / 128);
    gemm_nt_bf16<float><<<grid, 256, 0, stream>>>(attn, Wdb, out,
                                                  BATCH * SEQ, EMB, EMB);
  }
}
